// Attention_48369921687648
// MI455X (gfx1250) — matrix operand order repaired and checked by reading, not compiled, not run on hardware
//
#include <hip/hip_runtime.h>

typedef __attribute__((ext_vector_type(16))) _Float16 v16h;
typedef __attribute__((ext_vector_type(8)))  _Float16 v8h;
typedef __attribute__((ext_vector_type(4)))  _Float16 v4h;
typedef __attribute__((ext_vector_type(8)))  float    v8f;
typedef __attribute__((ext_vector_type(4)))  float    v4f;
typedef __attribute__((ext_vector_type(4)))  unsigned v4u;
typedef __attribute__((ext_vector_type(8)))  int      v8i;
typedef __attribute__((ext_vector_type(4)))  int      v4i;

#define B_  32
#define N_  512
#define E_  512
#define T_  2048
#define SA  528   // halves per weight-matrix row (512 + pad; rows are 16B multiples)
#define SB  40    // halves per K-chunk row in LDS (32 + 8 pad) == TDM pad output pitch

__constant__ constexpr float kEps         = 1e-6f;
__constant__ constexpr float kHalfLog2Pi  = 0.9189385332046727f;        // 0.5*ln(2*pi)
__constant__ constexpr float kNegLog1e4_E = -9.210340371976184f / 512.0f; // -ln(10000)/E

// ---------------------------------------------------------------------------
// Kernel A: per-(b,n) cumsum of durations -> mu, 1/sigma, additive constant.
// ---------------------------------------------------------------------------
__global__ void prep_kernel(const float* __restrict__ dur,
                            const float* __restrict__ rng,
                            float* __restrict__ mu_o,
                            float* __restrict__ is_o,
                            float* __restrict__ ca_o) {
  const int b = threadIdx.x;           // launched with exactly 32 threads
  const float* d = dur + b * N_;
  const float* r = rng + b * N_;
  float run = 0.0f;
  for (int n = 0; n < N_; ++n) {
    float dv = d[n];
    run += dv;
    float s = fmaxf(r[n], kEps);
    mu_o[b * N_ + n] = run - 0.5f * dv;
    is_o[b * N_ + n] = 1.0f / s;
    ca_o[b * N_ + n] = -__logf(s) - kHalfLog2Pi;
  }
}

// ---------------------------------------------------------------------------
// Kernel B: one-time f32 [b][n][e] -> f16 [b][e][n] transpose (LDS-tiled).
// Makes the WMMA B-operand K-contiguous so the TDM can feed it directly.
// ---------------------------------------------------------------------------
__global__ __launch_bounds__(256) void transpose_f16_kernel(
    const float* __restrict__ emb, _Float16* __restrict__ embT) {
  __shared__ float tile[64][68];
  const int b   = blockIdx.z;
  const int n0  = blockIdx.y * 64;
  const int e0  = blockIdx.x * 64;
  const int tid = threadIdx.x;

  const float* src = emb + ((size_t)b * N_ + n0) * E_ + e0;
  const int er = (tid & 15) * 4;
  const int nr = tid >> 4;
  #pragma unroll
  for (int i = 0; i < 4; ++i) {
    int n = nr + i * 16;
    v4f v = *(const v4f*)(src + (size_t)n * E_ + er);   // coalesced along e
    tile[n][er + 0] = v[0]; tile[n][er + 1] = v[1];
    tile[n][er + 2] = v[2]; tile[n][er + 3] = v[3];
  }
  __syncthreads();

  _Float16* dst = embT + ((size_t)b * E_ + e0) * N_ + n0;
  const int nw = (tid & 15) * 4;
  const int ew = tid >> 4;
  #pragma unroll
  for (int i = 0; i < 4; ++i) {
    int e = ew + i * 16;
    v4h h;
    h[0] = (_Float16)tile[nw + 0][e];
    h[1] = (_Float16)tile[nw + 1][e];
    h[2] = (_Float16)tile[nw + 2][e];
    h[3] = (_Float16)tile[nw + 3][e];
    *(v4h*)(dst + (size_t)e * N_ + nw) = h;             // coalesced along n
  }
}

// ---------------------------------------------------------------------------
// Kernel C: fused scores -> softmax(N) -> (T,N)@(N,E) WMMA -> +PE.
// Block = 256 threads (8 waves); covers 16 frames x all 512 channels.
// K-chunks of 32 are DMA'd by the Tensor Data Mover into double-buffered LDS.
// ---------------------------------------------------------------------------
__global__ __launch_bounds__(256) void fused_attn_kernel(
    const _Float16* __restrict__ embT,
    const float* __restrict__ mu_g,
    const float* __restrict__ is_g,
    const float* __restrict__ ca_g,
    float* __restrict__ out) {
  __shared__ __align__(16) _Float16 wA[16 * SA];        // softmax weights f16 [t][n]
  __shared__ __align__(16) _Float16 stage[2][E_ * SB];  // TDM dest: [e][k] pitch 40h
  __shared__ float red[16][17];
  __shared__ float rsum[16];

  const int tid = threadIdx.x;
  const int b   = blockIdx.y;
  const int t0  = blockIdx.x * 16;

  // ---------------- Phase 1: Gaussian scores + softmax over N --------------
  const int   t_loc = tid & 15;
  const int   sub   = tid >> 4;
  const int   nbase = sub * 32;
  const float tval  = (float)(t0 + t_loc);
  const float* mu_b = mu_g + b * N_;
  const float* is_b = is_g + b * N_;
  const float* ca_b = ca_g + b * N_;

  float lp[32];
  float lmax = -3.0e38f;
  #pragma unroll
  for (int j = 0; j < 32; ++j) {
    int n = nbase + j;
    float z = (tval - mu_b[n]) * is_b[n];
    float v = fmaf(-0.5f * z, z, ca_b[n]);
    lp[j] = v;
    lmax = fmaxf(lmax, v);
  }
  red[t_loc][sub] = lmax;
  __syncthreads();
  float m = red[t_loc][0];
  #pragma unroll
  for (int j = 1; j < 16; ++j) m = fmaxf(m, red[t_loc][j]);
  __syncthreads();

  float lsum = 0.0f;
  #pragma unroll
  for (int j = 0; j < 32; ++j) {
    float e = __expf(lp[j] - m);
    lsum += e;
    wA[t_loc * SA + nbase + j] = (_Float16)e;   // unnormalized; 1/sum in epilogue
  }
  red[t_loc][sub] = lsum;
  __syncthreads();
  if (sub == 0) {
    float s = 0.0f;
    #pragma unroll
    for (int j = 0; j < 16; ++j) s += red[t_loc][j];
    rsum[t_loc] = 1.0f / s;
  }
  // (first loop barrier publishes wA and rsum)

  // ---------------- Phase 2: TDM-pipelined WMMA contraction ----------------
  const int lane = tid & 31;
  const int wave = tid >> 5;
  const int half = lane >> 4;
  const int l15  = lane & 15;
  const _Float16* gbase = embT + (size_t)b * E_ * N_;   // [e][n] f16 slice

  // Issue a 2-D TDM load: tile 32(K, 64B contiguous) x 512(E rows, stride 512h),
  // with LDS padding 16B per 64B row -> rows land at the 80B (SB) pitch.
  auto issue_tdm = [&](int kc, int buf) {
    unsigned lds = (unsigned)(uintptr_t)(&stage[buf][0]);      // flat low 32b = LDS addr
    unsigned long long ga = (unsigned long long)(uintptr_t)(gbase + kc * 32);
    v4u g0 = { 1u,                                             // count=1 (valid D#)
               lds,                                            // lds_addr
               (unsigned)(ga & 0xFFFFFFFFu),                   // global_addr[31:0]
               (unsigned)((ga >> 32) & 0x1FFFFFFu) | (2u << 30) }; // ga[56:32] | type=2
    v8i g1 = { (int)((1u << 16) |        // data_size = 1 (2 bytes)
                     (1u << 20) |        // pad_enable
                     (3u << 22) |        // pad_interval: 16 DWORDs (= one 64B row)
                     (3u << 25)),        // pad_amount:   4 DWORDs (16B) -> 80B pitch
               (int)(512u << 16),        // tensor_dim0 = 512 (lo16 in bits 63:48)
               (int)(512u << 16),        // tensor_dim1 = 512 (lo16 in bits 95:80)
               (int)(32u  << 16),        // tile_dim0   = 32  (bits 127:112)
               512,                      // tile_dim1   = 512, tile_dim2 = 0
               512,                      // tensor_dim0_stride = 512 elements
               0, 0 };
    v4i gz4 = { 0, 0, 0, 0 };            // groups 2/3 unused (2-D tensor)
    v8i gz8 = { 0, 0, 0, 0, 0, 0, 0, 0 };
    // 6-arg form on this toolchain: (g0, g1, g2, g3, extra, cpol)
    __builtin_amdgcn_tensor_load_to_lds(g0, g1, gz4, gz4, gz8, 0);
  };

  v8f acc[4] = {};

  if (tid < 32) issue_tdm(0, 0);                 // prime the pipeline (wave 0)

  for (int kc = 0; kc < 16; ++kc) {
    if (tid < 32) {
      if (kc + 1 < 16) {
        issue_tdm(kc + 1, (kc + 1) & 1);         // prefetch next chunk via TDM
        __builtin_amdgcn_s_wait_tensorcnt((short)1);  // chunk kc has landed
      } else {
        __builtin_amdgcn_s_wait_tensorcnt((short)0);
      }
    }
    __syncthreads();                             // publish chunk kc (and wA on kc==0)

    // A fragment: lanes 0-15 -> K=kb..kb+15 (kb = half*16)
    const int ka = kc * 32 + half * 16;
    v8h alo = *(const v8h*)(&wA[l15 * SA + ka]);
    v8h ahi = *(const v8h*)(&wA[l15 * SA + ka + 8]);
    v16h a  = __builtin_shufflevector(alo, ahi,
                0,1,2,3,4,5,6,7,8,9,10,11,12,13,14,15);

    // Hoist all four B fragments (16 contiguous K per lane) before the WMMAs.
    const _Float16* bufp = &stage[kc & 1][0];
    const int kb = half * 16;
    v16h bfrag[4];
    #pragma unroll
    for (int tile = 0; tile < 4; ++tile) {
      int e = tile * 128 + wave * 16 + l15;
      v8h blo = *(const v8h*)(bufp + e * SB + kb);
      v8h bhi = *(const v8h*)(bufp + e * SB + kb + 8);
      bfrag[tile] = __builtin_shufflevector(blo, bhi,
                      0,1,2,3,4,5,6,7,8,9,10,11,12,13,14,15);
    }
    #pragma unroll
    for (int tile = 0; tile < 4; ++tile) {
      acc[tile] = __builtin_amdgcn_wmma_f32_16x16x32_f16(
          false, a, false, bfrag[tile], (short)0, acc[tile], false, false);
    }
    __syncthreads();       // chunk kc consumed; its buffer may be overwritten next
  }

  // ---------------- Phase 3: normalize + positional encoding + store -------
  float* outB = out + (size_t)b * T_ * E_;
  #pragma unroll
  for (int tile = 0; tile < 4; ++tile) {
    int e = tile * 128 + wave * 16 + l15;
    float fe   = (float)(e & ~1);
    float freq = __expf(fe * kNegLog1e4_E);
    #pragma unroll
    for (int r2 = 0; r2 < 8; ++r2) {
      int mrow = r2 + half * 8;        // C layout: lanes 0-15 M=r, lanes 16-31 M=8+r
      int t    = t0 + mrow;
      float ang = (float)t * freq;
      float pe  = (e & 1) ? __cosf(ang) : __sinf(ang);
      float val = acc[tile][r2] * rsum[mrow] + pe;
      outB[(size_t)t * E_ + e] = val;
    }
  }
}

// ---------------------------------------------------------------------------
// Workspace layout (requires ws_size >= ~17 MB):
//   [0)                 f16 embT  : B*E*N halves  = 16,777,216 bytes
//   [embT_end)          f32 mu    : B*N
//   ...                 f32 inv_s : B*N
//   ...                 f32 cadd  : B*N
// ---------------------------------------------------------------------------
extern "C" void kernel_launch(void* const* d_in, const int* in_sizes, int n_in,
                              void* d_out, int out_size, void* d_ws, size_t ws_size,
                              hipStream_t stream) {
  (void)in_sizes; (void)n_in; (void)out_size; (void)ws_size;
  const float* emb = (const float*)d_in[0];
  const float* dur = (const float*)d_in[1];
  const float* rng = (const float*)d_in[2];
  // d_in[3] is the scalar T (=2048), compiled in as a constant.

  _Float16* embT = (_Float16*)d_ws;
  float* fws = (float*)((char*)d_ws + (size_t)B_ * E_ * N_ * sizeof(_Float16));
  float* mu  = fws;
  float* is  = fws +     B_ * N_;
  float* ca  = fws + 2 * B_ * N_;

  transpose_f16_kernel<<<dim3(E_ / 64, N_ / 64, B_), 256, 0, stream>>>(emb, embT);
  prep_kernel<<<1, 32, 0, stream>>>(dur, rng, mu, is, ca);

  fused_attn_kernel<<<dim3(T_ / 16, B_), 256, 0, stream>>>(
      embT, mu, is, ca, (float*)d_out);
}
